// BooleanAnchorTable_32899449487397
// MI455X (gfx1250) — matrix operand order repaired and checked by reading, not compiled, not run on hardware
//
#include <hip/hip_runtime.h>

// ---------------------------------------------------------------------------
// VQ codebook lookup for MI455X (gfx1250): BF16 WMMA distance matmul + argmin
// Double-buffered GLOBAL_LOAD_ASYNC_TO_LDS staging (ASYNCcnt) + WMMA pipeline
// ---------------------------------------------------------------------------

typedef __bf16 bf16_t;
typedef bf16_t v16bf __attribute__((ext_vector_type(16)));
typedef float  v8f   __attribute__((ext_vector_type(8)));
typedef unsigned int u32x4 __attribute__((ext_vector_type(4)));

#define HID     512
#define CODES   8192
#define TOKENS  8192
#define MB      64                 // tokens per block
#define NB      64                 // codes per N-chunk (double-buffered)
#define LDT     520                // padded LDS row stride (bf16); 1040B = 65*16B
#define NCHUNK  (CODES / NB)       // 128
#define KSTEPS  (HID / 32)         // 16 WMMA K-steps
#define IDXOFF  (TOKENS * HID)     // 4194304
#define LOSS0   (IDXOFF + TOKENS)  // 4202496
#define LOSS1   (LOSS0 + 1)
#define LSCALE  (1.0f / 4194304.0f)

#if defined(__gfx1250__)
#define WAIT_ASYNC(n) asm volatile("s_wait_asynccnt %0" :: "i"(n) : "memory")
#else
#define WAIT_ASYNC(n)
#endif

union FragBF { u32x4 q[2]; v16bf v; };

static __device__ __forceinline__ unsigned short f2bf(float f) {
    unsigned int u = __float_as_uint(f);
    u += 0x7FFFu + ((u >> 16) & 1u);          // round-to-nearest-even
    return (unsigned short)(u >> 16);
}

// Async global->LDS 16B transfer (tracked by ASYNCcnt); fallback: plain copy.
static __device__ __forceinline__ void ld_async_b128(u32x4* lds_dst,
                                                     const u32x4* gsrc) {
#if defined(__gfx1250__)
    unsigned lds = (unsigned)(uintptr_t)lds_dst;                 // LDS offset
    unsigned long long ga = (unsigned long long)(uintptr_t)gsrc; // 64b vaddr
    asm volatile("global_load_async_to_lds_b128 %0, %1, off"
                 :: "v"(lds), "v"(ga) : "memory");
#else
    *lds_dst = *gsrc;
#endif
}

// --- prep: z (f32) -> bf16 ------------------------------------------------
__global__ __launch_bounds__(256) void prep_z(const float* __restrict__ z,
                                              unsigned short* __restrict__ zb) {
    int gid = blockIdx.x * 256 + threadIdx.x;
    int base = gid * 4;
#pragma unroll
    for (int i = 0; i < 4; ++i) zb[base + i] = f2bf(z[base + i]);
}

// --- prep: emb (f32) -> bf16, ||e||^2, zero loss slots --------------------
__global__ __launch_bounds__(256) void prep_emb(const float* __restrict__ emb,
                                                unsigned short* __restrict__ eb,
                                                float* __restrict__ e2,
                                                float* __restrict__ out) {
    __shared__ float sp[8];
    int row = blockIdx.x;
    int tid = threadIdx.x;
    const float* er = emb + (size_t)row * HID;
    float s = 0.f;
#pragma unroll
    for (int j = tid; j < HID; j += 256) {
        float v = er[j];
        eb[(size_t)row * HID + j] = f2bf(v);
        s += v * v;
    }
#pragma unroll
    for (int off = 16; off; off >>= 1) s += __shfl_xor(s, off, 32);
    if ((tid & 31) == 0) sp[tid >> 5] = s;
    __syncthreads();
    if (tid == 0) {
        float t = 0.f;
#pragma unroll
        for (int i = 0; i < 8; ++i) t += sp[i];
        e2[row] = t;
        if (row == 0) { out[LOSS0] = 0.f; out[LOSS1] = 0.f; }
    }
}

// --- main: WMMA distances + argmin + gather + losses ----------------------
__global__ __launch_bounds__(256) void vq_main(const float* __restrict__ zf,
                                               const float* __restrict__ embf,
                                               const unsigned short* __restrict__ zb,
                                               const unsigned short* __restrict__ eb,
                                               const float* __restrict__ e2,
                                               float* __restrict__ out) {
    extern __shared__ unsigned short dsm[];                // dynamic LDS
    unsigned short* sA  = dsm;                             // 64 x LDT bf16
    unsigned short* sB0 = dsm + MB * LDT;                  // 64 x LDT bf16
    unsigned short* sB1 = sB0 + NB * LDT;                  // 64 x LDT bf16
    __shared__ float s_minv[2][MB];
    __shared__ int   s_mini[2][MB];
    __shared__ int   s_idx[MB];
    __shared__ float s_red[8];

    const int tid  = threadIdx.x;
    const int lane = tid & 31;
    const int w    = tid >> 5;            // wave 0..7
    const int m0   = (w & 3) * 16;        // wave's token sub-tile
    const int ng   = w >> 2;              // N half (0 or 1): 32 codes each
    const int half = lane >> 4;
    const int l15  = lane & 15;

    // ---- async staging helpers (16 x b128 per thread each) --------------
    auto stageA = [&]() {
        const u32x4* src = (const u32x4*)(zb + (size_t)blockIdx.x * MB * HID);
#pragma unroll
        for (int i = 0; i < 16; ++i) {
            int q = tid + i * 256;
            int row = q >> 6, col = q & 63;
            ld_async_b128((u32x4*)(sA + row * LDT) + col, src + row * 64 + col);
        }
    };
    auto stageB = [&](unsigned short* buf, int cc) {
        const u32x4* src = (const u32x4*)(eb + (size_t)cc * NB * HID);
#pragma unroll
        for (int i = 0; i < 16; ++i) {
            int q = tid + i * 256;
            int row = q >> 6, col = q & 63;
            ld_async_b128((u32x4*)(buf + row * LDT) + col, src + row * 64 + col);
        }
    };

    stageA();              // 16 outstanding async (A tile)
    stageB(sB0, 0);        // +16 (chunk 0)

    float minv[8];
    int   mini[8];
#pragma unroll
    for (int r = 0; r < 8; ++r) { minv[r] = __builtin_inff(); mini[r] = 0; }

    const unsigned short* aRow = sA + (size_t)(m0 + l15) * LDT;

    for (int c = 0; c < NCHUNK; ++c) {
        if (c + 1 < NCHUNK) {
            stageB((c & 1) ? sB0 : sB1, c + 1);  // prefetch next chunk
            WAIT_ASYNC(16);   // retire A + chunk c (async loads done in-order)
        } else {
            WAIT_ASYNC(0);
        }
        __syncthreads();      // chunk c visible to all waves

        const unsigned short* bBase = (c & 1) ? sB1 : sB0;
        const unsigned short* bRow0 = bBase + (size_t)(ng * 32 + l15) * LDT;
        const unsigned short* bRow1 = bRow0 + (size_t)16 * LDT;

        v8f acc0 = {}, acc1 = {};
#pragma unroll
        for (int ks = 0; ks < KSTEPS; ++ks) {
            const int kb = ks * 32;
            FragBF a, b0, b1;
            // A 16x32 bf16: lane half selects contiguous K {h*16 .. h*16+15}
            a.q[0]  = *(const u32x4*)(aRow + kb + half * 16);
            a.q[1]  = *(const u32x4*)(aRow + kb + half * 16 + 8);
            // B 32x16 bf16: lane half selects contiguous K {h*16 .. h*16+15}
            b0.q[0] = *(const u32x4*)(bRow0 + kb + half * 16);
            b0.q[1] = *(const u32x4*)(bRow0 + kb + half * 16 + 8);
            b1.q[0] = *(const u32x4*)(bRow1 + kb + half * 16);
            b1.q[1] = *(const u32x4*)(bRow1 + kb + half * 16 + 8);
            acc0 = __builtin_amdgcn_wmma_f32_16x16x32_bf16(
                false, a.v, false, b0.v, (short)0, acc0, false, false);
            acc1 = __builtin_amdgcn_wmma_f32_16x16x32_bf16(
                false, a.v, false, b1.v, (short)0, acc1, false, false);
        }

        // score = ||e||^2 - 2 x.e  (||x||^2 constant per token for argmin)
        const int gn0 = c * NB + ng * 32 + l15;
        const float e20 = e2[gn0];
        const float e21 = e2[gn0 + 16];
#pragma unroll
        for (int r = 0; r < 8; ++r) {
            float sc0 = __builtin_fmaf(-2.0f, acc0[r], e20);
            float sc1 = __builtin_fmaf(-2.0f, acc1[r], e21);
            if (sc0 < minv[r]) { minv[r] = sc0; mini[r] = gn0; }
            if (sc1 < minv[r]) { minv[r] = sc1; mini[r] = gn0 + 16; }
        }
        __syncthreads();      // done with buf[c&1] before it is re-staged
    }

    // cross-lane argmin within each 16-lane half (C layout: M = r + 8*half)
#pragma unroll
    for (int r = 0; r < 8; ++r) {
        float v = minv[r];
        int   ix = mini[r];
#pragma unroll
        for (int off = 1; off < 16; off <<= 1) {
            float ov = __shfl_xor(v, off, 32);
            int   oi = __shfl_xor(ix, off, 32);
            if (ov < v || (ov == v && oi < ix)) { v = ov; ix = oi; }
        }
        if (l15 == 0) {
            int t = m0 + r + 8 * half;
            s_minv[ng][t] = v;
            s_mini[ng][t] = ix;
        }
    }
    __syncthreads();

    if (tid < MB) {
        float v0 = s_minv[0][tid], v1 = s_minv[1][tid];
        int   i0 = s_mini[0][tid], i1 = s_mini[1][tid];
        int best = (v1 < v0 || (v1 == v0 && i1 < i0)) ? i1 : i0;
        s_idx[tid] = best;
        out[IDXOFF + blockIdx.x * MB + tid] = (float)best;
    }
    __syncthreads();

    // gather z_q rows (exact f32 codebook values) + fused loss accumulation
    float lsum = 0.f;
    for (int t = 0; t < MB; ++t) {
        const int ci = s_idx[t];
        const float* er = embf + (size_t)ci * HID;
        const size_t tok = (size_t)blockIdx.x * MB + t;
        const float* zr = zf + tok * HID;
        float* orow = out + tok * HID;
#pragma unroll
        for (int j = tid; j < HID; j += 256) {
            float q = er[j];
            float d = q - zr[j];
            orow[j] = q;           // z_st == z_q in forward value
            lsum += d * d;
        }
    }
#pragma unroll
    for (int off = 16; off; off >>= 1) lsum += __shfl_xor(lsum, off, 32);
    if (lane == 0) s_red[w] = lsum;
    __syncthreads();
    if (tid == 0) {
        float tot = 0.f;
#pragma unroll
        for (int i = 0; i < 8; ++i) tot += s_red[i];
        float contrib = tot * LSCALE;
        atomicAdd(out + LOSS0, contrib);   // codebook loss
        atomicAdd(out + LOSS1, contrib);   // commit loss (numerically equal)
    }
}

extern "C" void kernel_launch(void* const* d_in, const int* in_sizes, int n_in,
                              void* d_out, int out_size, void* d_ws, size_t ws_size,
                              hipStream_t stream) {
    const float* z   = (const float*)d_in[0];   // (8,1024,512) f32
    const float* emb = (const float*)d_in[1];   // (8192,512)   f32
    float* out = (float*)d_out;

    unsigned short* zb = (unsigned short*)d_ws;            // 8 MB bf16 z
    unsigned short* eb = zb + (size_t)TOKENS * HID;        // 8 MB bf16 emb
    float* e2 = (float*)(eb + (size_t)CODES * HID);        // 32 KB ||e||^2

    prep_z<<<(TOKENS * HID) / (256 * 4), 256, 0, stream>>>(z, zb);
    prep_emb<<<CODES, 256, 0, stream>>>(emb, eb, e2, out);

    const size_t dyn = (size_t)(MB + 2 * NB) * LDT * sizeof(unsigned short); // 199680 B
    vq_main<<<TOKENS / MB, 256, dyn, stream>>>(z, emb, zb, eb, e2, out);
}
